// DCNv3Conv_58935541236114
// MI455X (gfx1250) — matrix operand order repaired and checked by reading, not compiled, not run on hardware
//
#include <hip/hip_runtime.h>
#include <hip/hip_bf16.h>
#include <math.h>

#define BB   4
#define CC   512
#define HH   56
#define WW   56
#define GG   16
#define PPK  9
#define GCC  32
#define HWSZ (HH * WW)          // 3136
#define NPIX (BB * HWSZ)        // 12544
#define NOFF (GG * PPK * 2)     // 288
#define NOM  (GG * PPK * 3)     // 432 = offsets(288) + mask(144)

typedef __attribute__((ext_vector_type(16))) _Float16 v16h;
typedef __attribute__((ext_vector_type(8)))  _Float16 v8h;
typedef __attribute__((ext_vector_type(8)))  float    v8f;

union AFrag { v16h v; v8h h[2]; };

// ---------------------------------------------------------------------------
// 1) NCHW fp32 -> NHWC fp32 + NHWC f16
// ---------------------------------------------------------------------------
__global__ void k_transpose_in(const float* __restrict__ x,
                               float* __restrict__ xh32,
                               _Float16* __restrict__ xh16) {
    int idx = blockIdx.x * 256 + threadIdx.x;       // over NPIX*CC (exact)
    int c   = idx & (CC - 1);
    int pix = idx >> 9;
    int b   = pix / HWSZ;
    int hw  = pix - b * HWSZ;
    float v = x[((size_t)(b * CC + c)) * HWSZ + hw];
    xh32[idx] = v;
    xh16[idx] = (_Float16)v;
}

// ---------------------------------------------------------------------------
// 2) weight [K, Nc] fp32 -> transposed f16 [rowOff + Nc, K]
// ---------------------------------------------------------------------------
__global__ void k_convert_wT(const float* __restrict__ w, _Float16* __restrict__ wT,
                             int K, int Nc, int rowOff) {
    int idx = blockIdx.x * 256 + threadIdx.x;
    if (idx >= K * Nc) return;
    int n = idx / K;
    int k = idx - n * K;
    wT[(size_t)(rowOff + n) * K + k] = (_Float16)w[(size_t)k * Nc + n];
}

__global__ void k_cat_bias(const float* __restrict__ boff,
                           const float* __restrict__ bmask,
                           float* __restrict__ bcat) {
    int i = blockIdx.x * 256 + threadIdx.x;
    if (i < NOFF)      bcat[i] = boff[i];
    else if (i < NOM)  bcat[i] = bmask[i - NOFF];
}

// ---------------------------------------------------------------------------
// Async A-tile stage: 256 threads each move one 16B chunk of the 64x32 f16
// tile (4KB) global -> LDS via the CDNA5 async LDS-DMA path (ASYNCcnt).
// ---------------------------------------------------------------------------
__device__ __forceinline__ void issue_a_async(const _Float16* __restrict__ A,
                                              int m0, int K, int kstep,
                                              unsigned ldsbase, int buf) {
    const int ci   = threadIdx.x;     // 0..255
    const int row  = ci >> 2;         // 0..63
    const int part = ci & 3;          // 4 x 16B per 64B row
    const _Float16* g = A + (size_t)(m0 + row) * K + kstep * 32 + part * 8;
    unsigned laddr = ldsbase + (unsigned)(buf * 4096 + row * 64 + part * 16);
    asm volatile("global_load_async_to_lds_b128 %0, %1, off"
                 :: "v"(laddr), "v"((unsigned long long)(uintptr_t)g)
                 : "memory");
}

// ---------------------------------------------------------------------------
// Shared GEMM core: acc[4] = A[m0:m0+64, :K] * B[:, nt*16:+16]  (f16 WMMA)
// A double-buffered through LDS; B streamed from global (per-wave column).
// ALL waves of the 256-thread block must call this (block barriers inside).
// ---------------------------------------------------------------------------
__device__ __forceinline__ void gemm_core(const _Float16* __restrict__ A,
                                          const _Float16* __restrict__ BT,
                                          int m0, int ntc, int K,
                                          _Float16* sA, v8f acc[4]) {
    const unsigned ldsbase = (unsigned)(uintptr_t)sA;
    const int lane = threadIdx.x & 31;
    const int hl   = lane >> 4;
    const int ln   = lane & 15;
    const _Float16* bptr = BT + (size_t)(ntc * 16 + ln) * K + 16 * hl;
    const _Float16* s0   = sA + ln * 32 + hl * 16;

    issue_a_async(A, m0, K, 0, ldsbase, 0);
    const int niter = K / 32;
    for (int i = 0; i < niter; ++i) {
        const int cur = i & 1;
        if (i + 1 < niter) {
            issue_a_async(A, m0, K, i + 1, ldsbase, (i + 1) & 1);
            asm volatile("s_wait_asynccnt 0x1" ::: "memory");
        } else {
            asm volatile("s_wait_asynccnt 0x0" ::: "memory");
        }
        __syncthreads();                       // A tile buf[cur] visible to all

        v16h bf = *(const v16h*)(bptr + i * 32);
        const _Float16* sc = s0 + cur * 2048;

        // hoist all LDS fragment reads so one dscnt wait covers the
        // whole group and the 4 WMMAs issue back-to-back
        AFrag a[4];
#pragma unroll
        for (int t = 0; t < 4; ++t) {
            a[t].h[0] = *(const v8h*)(sc + t * 16 * 32);
            a[t].h[1] = *(const v8h*)(sc + t * 16 * 32 + 8);
        }
#pragma unroll
        for (int t = 0; t < 4; ++t) {
            acc[t] = __builtin_amdgcn_wmma_f32_16x16x32_f16(
                false, a[t].v, false, bf, (short)0, acc[t], false, false);
        }
        __syncthreads();                       // reads done before refill
    }
}

// ---------------------------------------------------------------------------
// 3) generic f16 WMMA GEMM: C[M,Nc] = A[M,K] * BT^T + bias, fp32 out.
// ---------------------------------------------------------------------------
__global__ void k_gemm_f16(const _Float16* __restrict__ A,
                           const _Float16* __restrict__ BT,
                           const float* __restrict__ bias,
                           float* __restrict__ Co,
                           int M, int Nc, int K) {
    __shared__ _Float16 sA[2 * 64 * 32];
    const int lane = threadIdx.x & 31;
    const int wv   = threadIdx.x >> 5;
    const int hl   = lane >> 4;
    const int ln   = lane & 15;
    const int nt   = blockIdx.y * 8 + wv;
    const int ntc  = (nt * 16 < Nc) ? nt : (Nc / 16 - 1);   // clamp, keep barriers uniform
    const int m0   = blockIdx.x * 64;

    v8f acc[4] = {};
    gemm_core(A, BT, m0, ntc, K, sA, acc);

    if (nt * 16 >= Nc) return;
    const int col = nt * 16 + ln;
    const float bsv = bias[col];
#pragma unroll
    for (int t = 0; t < 4; ++t) {
#pragma unroll
        for (int i = 0; i < 8; ++i) {
            int row = m0 + t * 16 + hl * 8 + i;
            Co[(size_t)row * Nc + col] = acc[t][i] + bsv;
        }
    }
}

// ---------------------------------------------------------------------------
// 4) depthwise 3x3 + LayerNorm(C) + exact GELU -> f16
// ---------------------------------------------------------------------------
__global__ void k_dw_ln_gelu(const float* __restrict__ xh,
                             const float* __restrict__ w_dw,
                             const float* __restrict__ b_dw,
                             const float* __restrict__ ln_g,
                             const float* __restrict__ ln_b,
                             _Float16* __restrict__ x1) {
    const int pix = blockIdx.x;
    const int b  = pix / HWSZ;
    const int hw = pix - b * HWSZ;
    const int h  = hw / WW;
    const int w  = hw - h * WW;
    const int tid = threadIdx.x;

    float acc[2];
#pragma unroll
    for (int j = 0; j < 2; ++j) {
        int c = tid + j * 256;
        float a = b_dw[c];
#pragma unroll
        for (int ky = 0; ky < 3; ++ky) {
            int hhh = h + ky - 1;
            if (hhh < 0 || hhh >= HH) continue;
#pragma unroll
            for (int kx = 0; kx < 3; ++kx) {
                int www = w + kx - 1;
                if (www < 0 || www >= WW) continue;
                a += xh[((size_t)((b * HH + hhh) * WW + www)) * CC + c]
                     * w_dw[(ky * 3 + kx) * CC + c];
            }
        }
        acc[j] = a;
    }

    __shared__ float rs[256];
    __shared__ float rq[256];
    rs[tid] = acc[0] + acc[1];
    rq[tid] = acc[0] * acc[0] + acc[1] * acc[1];
    __syncthreads();
    for (int off = 128; off > 0; off >>= 1) {
        if (tid < off) { rs[tid] += rs[tid + off]; rq[tid] += rq[tid + off]; }
        __syncthreads();
    }
    const float mean = rs[0] * (1.0f / CC);
    const float var  = rq[0] * (1.0f / CC) - mean * mean;
    const float inv  = rsqrtf(var + 1e-6f);

#pragma unroll
    for (int j = 0; j < 2; ++j) {
        int c = tid + j * 256;
        float v = (acc[j] - mean) * inv * ln_g[c] + ln_b[c];
        v = 0.5f * v * (1.0f + erff(v * 0.70710678118654752f));
        x1[(size_t)pix * CC + c] = (_Float16)v;
    }
}

// ---------------------------------------------------------------------------
// 5) softmax over P=9 for mask portion of offmask buffer (in-place)
// ---------------------------------------------------------------------------
__global__ void k_softmax_mask(float* __restrict__ om) {
    int idx = blockIdx.x * 256 + threadIdx.x;     // over NPIX*GG (exact)
    int row = idx >> 4;
    int g   = idx & 15;
    float* p = om + (size_t)row * NOM + NOFF + g * PPK;
    float mx = p[0];
#pragma unroll
    for (int i = 1; i < PPK; ++i) mx = fmaxf(mx, p[i]);
    float e[PPK];
    float s = 0.f;
#pragma unroll
    for (int i = 0; i < PPK; ++i) { e[i] = expf(p[i] - mx); s += e[i]; }
    float invs = 1.0f / s;
#pragma unroll
    for (int i = 0; i < PPK; ++i) p[i] = e[i] * invs;
}

// ---------------------------------------------------------------------------
// 6) DCNv3 bilinear gather core. One wave per (pixel, group); lane = channel.
// ---------------------------------------------------------------------------
__device__ __forceinline__ float dcn_fetch(const float* __restrict__ base, int yi, int xi) {
    if (yi < 1 || yi > HH || xi < 1 || xi > WW) return 0.f;
    return base[(size_t)((yi - 1) * WW + (xi - 1)) * CC];
}

__global__ void k_dcnv3(const float* __restrict__ xp,
                        const float* __restrict__ om,
                        _Float16* __restrict__ y) {
    const int lane = threadIdx.x & 31;
    const int wv   = threadIdx.x >> 5;
    const int widx = blockIdx.x * 8 + wv;        // 0 .. NPIX*GG-1
    const int g    = widx & 15;
    const int row  = widx >> 4;
    const int b    = row / HWSZ;
    const int hw   = row - b * HWSZ;
    const int h    = hw / WW;
    const int w    = hw - h * WW;

    const float* offp = om + (size_t)row * NOM + g * (PPK * 2);
    const float* mp   = om + (size_t)row * NOM + NOFF + g * PPK;
    const float* base = xp + ((size_t)b * HWSZ) * CC + g * GCC + lane;

    float acc = 0.f;
#pragma unroll
    for (int p = 0; p < PPK; ++p) {
        const int kx = p / 3 - 1;        // repeat(k, K)
        const int ky = p % 3 - 1;        // tile(k, K)
        float px = (float)(w + 1 + kx) + offp[2 * p + 0];
        float py = (float)(h + 1 + ky) + offp[2 * p + 1];
        float x0f = floorf(px), y0f = floorf(py);
        float tx = px - x0f, ty = py - y0f;
        int xi = (int)x0f, yi = (int)y0f;
        float c00 = dcn_fetch(base, yi,     xi);
        float c01 = dcn_fetch(base, yi,     xi + 1);
        float c10 = dcn_fetch(base, yi + 1, xi);
        float c11 = dcn_fetch(base, yi + 1, xi + 1);
        float bl = (1.f - ty) * ((1.f - tx) * c00 + tx * c01)
                 +        ty  * ((1.f - tx) * c10 + tx * c11);
        acc += mp[p] * bl;
    }
    y[(size_t)row * CC + g * GCC + lane] = (_Float16)acc;
}

// ---------------------------------------------------------------------------
// 7) output proj GEMM + BN(inference) + SiLU + NHWC->NCHW scatter
// ---------------------------------------------------------------------------
__global__ void k_gemm_out(const _Float16* __restrict__ A,
                           const _Float16* __restrict__ BT,
                           const float* __restrict__ b_out,
                           const float* __restrict__ bn_g,
                           const float* __restrict__ bn_b,
                           const float* __restrict__ bn_m,
                           const float* __restrict__ bn_v,
                           float* __restrict__ out) {
    __shared__ _Float16 sA[2 * 64 * 32];
    const int lane = threadIdx.x & 31;
    const int wv   = threadIdx.x >> 5;
    const int hl   = lane >> 4;
    const int ln   = lane & 15;
    const int nt   = blockIdx.y * 8 + wv;        // Nc = 512 -> always valid
    const int m0   = blockIdx.x * 64;

    v8f acc[4] = {};
    gemm_core(A, BT, m0, nt, CC, sA, acc);

    const int col = nt * 16 + ln;                // channel
    const float bo = b_out[col];
    const float sc = bn_g[col] * rsqrtf(bn_v[col] + 1e-5f);
    const float sh = bn_b[col] - bn_m[col] * sc;
#pragma unroll
    for (int t = 0; t < 4; ++t) {
#pragma unroll
        for (int i = 0; i < 8; ++i) {
            int row = m0 + t * 16 + hl * 8 + i;
            int b   = row / HWSZ;
            int hw  = row - b * HWSZ;
            float v = acc[t][i] + bo;
            v = v * sc + sh;
            v = v / (1.0f + expf(-v));           // SiLU
            out[((size_t)(b * CC + col)) * HWSZ + hw] = v;
        }
    }
}

// ---------------------------------------------------------------------------
// host launcher
// ---------------------------------------------------------------------------
extern "C" void kernel_launch(void* const* d_in, const int* in_sizes, int n_in,
                              void* d_out, int out_size, void* d_ws, size_t ws_size,
                              hipStream_t stream) {
    const float* x      = (const float*)d_in[0];
    const float* w_in   = (const float*)d_in[1];
    const float* b_in   = (const float*)d_in[2];
    const float* w_dw   = (const float*)d_in[3];
    const float* b_dw   = (const float*)d_in[4];
    const float* ln_g   = (const float*)d_in[5];
    const float* ln_b   = (const float*)d_in[6];
    const float* w_off  = (const float*)d_in[7];
    const float* b_off  = (const float*)d_in[8];
    const float* w_mask = (const float*)d_in[9];
    const float* b_mask = (const float*)d_in[10];
    const float* w_out  = (const float*)d_in[11];
    const float* b_out  = (const float*)d_in[12];
    const float* bn_g   = (const float*)d_in[13];
    const float* bn_b   = (const float*)d_in[14];
    const float* bn_m   = (const float*)d_in[15];
    const float* bn_v   = (const float*)d_in[16];
    float* out = (float*)d_out;

    char* ws = (char*)d_ws;
    const size_t SZ_NC16 = (size_t)NPIX * CC * sizeof(_Float16);
    const size_t SZ_NC32 = (size_t)NPIX * CC * sizeof(float);
    _Float16* xh16   = (_Float16*)(ws);                            size_t o = SZ_NC16;
    float*    xh32   = (float*)(ws + o);                           o += SZ_NC32;
    float*    xp     = (float*)(ws + o);                           o += SZ_NC32;
    _Float16* x1_16  = (_Float16*)(ws + o);                        o += SZ_NC16;
    _Float16* y16    = (_Float16*)(ws + o);                        o += SZ_NC16;
    float*    offm   = (float*)(ws + o);                           o += (size_t)NPIX * NOM * sizeof(float);
    _Float16* w_inT  = (_Float16*)(ws + o);                        o += (size_t)CC * CC * sizeof(_Float16);
    _Float16* w_omT  = (_Float16*)(ws + o);                        o += (size_t)NOM * CC * sizeof(_Float16);
    _Float16* w_outT = (_Float16*)(ws + o);                        o += (size_t)CC * CC * sizeof(_Float16);
    float*    b_cat  = (float*)(ws + o);                           o += 512 * sizeof(float);

    // 1) layout + precision prep
    k_transpose_in<<<(NPIX * CC) / 256, 256, 0, stream>>>(x, xh32, xh16);
    k_convert_wT<<<(CC * CC + 255) / 256, 256, 0, stream>>>(w_in,   w_inT,  CC, CC,   0);
    k_convert_wT<<<(CC * NOFF + 255) / 256, 256, 0, stream>>>(w_off,  w_omT,  CC, NOFF, 0);
    k_convert_wT<<<(CC * (NOM - NOFF) + 255) / 256, 256, 0, stream>>>(w_mask, w_omT, CC, NOM - NOFF, NOFF);
    k_convert_wT<<<(CC * CC + 255) / 256, 256, 0, stream>>>(w_out,  w_outT, CC, CC,   0);
    k_cat_bias<<<2, 256, 0, stream>>>(b_off, b_mask, b_cat);

    const dim3 gemm_grid(NPIX / 64, 4);   // 196 M-blocks x (8 waves x 4) N-tiles

    // 2) input projection: xp = xh @ w_in + b_in
    k_gemm_f16<<<gemm_grid, 256, 0, stream>>>(xh16, w_inT, b_in, xp, NPIX, CC, CC);

    // 3) depthwise + LN + GELU
    k_dw_ln_gelu<<<NPIX, 256, 0, stream>>>(xh32, w_dw, b_dw, ln_g, ln_b, x1_16);

    // 4) fused offset|mask GEMM (Nc = 432), then softmax over P
    k_gemm_f16<<<gemm_grid, 256, 0, stream>>>(x1_16, w_omT, b_cat, offm, NPIX, NOM, CC);
    k_softmax_mask<<<(NPIX * GG) / 256, 256, 0, stream>>>(offm);

    // 5) deformable bilinear gather core
    k_dcnv3<<<(NPIX * GG) / 8, 256, 0, stream>>>(xp, offm, y16);

    // 6) output projection + BN + SiLU + NCHW scatter
    k_gemm_out<<<gemm_grid, 256, 0, stream>>>(y16, w_outT, b_out, bn_g, bn_b, bn_m, bn_v, out);
}
